// CrossImageAttentionLayer_89446988907041
// MI455X (gfx1250) — matrix operand order repaired and checked by reading, not compiled, not run on hardware
//
#include <hip/hip_runtime.h>

// ---------- types for CDNA5 WMMA / TDM ----------
typedef __attribute__((ext_vector_type(16))) __bf16        v16bf;
typedef __attribute__((ext_vector_type(8)))  float         v8f;
typedef __attribute__((ext_vector_type(4)))  unsigned int  u32x4;
typedef __attribute__((ext_vector_type(8)))  int           i32x8;
typedef __attribute__((ext_vector_type(4)))  int           i32x4;

#if defined(__has_builtin)
#if __has_builtin(__builtin_amdgcn_tensor_load_to_lds)
#define USE_TDM 1
#endif
#endif
#ifndef USE_TDM
#define USE_TDM 0
#endif

// ---------- helpers ----------
__device__ __forceinline__ unsigned short f2bf(float f) {
    union { float f; unsigned int u; } c; c.f = f;
    unsigned int u = c.u;
    u += 0x7fffu + ((u >> 16) & 1u);      // round-to-nearest-even
    return (unsigned short)(u >> 16);
}
__device__ __forceinline__ float bf2f(unsigned short h) {
    union { unsigned int u; float f; } c; c.u = ((unsigned int)h) << 16;
    return c.f;
}
__device__ __forceinline__ unsigned int pack2bf(float a, float b) {
    return (unsigned int)f2bf(a) | ((unsigned int)f2bf(b) << 16);
}
__device__ __forceinline__ float waveSum(float v) {
    #pragma unroll
    for (int off = 16; off > 0; off >>= 1) v += __shfl_xor(v, off, 32);
    return v;   // all 32 lanes hold the sum (wave32)
}

#if USE_TDM
// TDM: copy 32 rows x 512B (contiguous 16 KiB) from global into LDS, inserting a
// 4-dword pad after every 128 dwords -> LDS row stride 132 dwords (matches the
// padded layout the WMMA fragment reads expect). D# per CDNA5 ISA §8.3/8.4.
__device__ __forceinline__ void tdm_load_a_tile(const void* gsrc, void* ldsDst) {
    unsigned long long ga = (unsigned long long)gsrc;
    unsigned int lds = (unsigned int)(unsigned long long)ldsDst; // addr[31:0] = LDS offset
    constexpr unsigned ELEMS = 32u * 512u / 8u;  // 2048 x 8-byte elements
    u32x4 g0;
    g0[0] = 1u;                                   // count=1 (user descriptor)
    g0[1] = lds;                                  // lds_addr
    g0[2] = (unsigned int)(ga & 0xffffffffu);     // global_addr[31:0]
    g0[3] = (unsigned int)((ga >> 32) & 0x01ffffffu) | (2u << 30);  // addr[56:32] | type=2
    i32x8 g1;
    // data_size=3 (8B) | pad_enable | pad_interval=6 (128 dw) | pad_amount=3 (4 dw)
    g1[0] = (int)((3u << 16) | (1u << 20) | (6u << 22) | (3u << 25));
    g1[1] = (int)((ELEMS & 0xffffu) << 16);                    // tensor_dim0[15:0] @ bits63:48
    g1[2] = (int)(((ELEMS >> 16) & 0xffffu) | (1u << 16));     // tensor_dim0 hi | tensor_dim1=1
    g1[3] = (int)((ELEMS & 0xffffu) << 16);                    // tile_dim0 @ bits127:112
    g1[4] = 1;                                                 // tile_dim1=1
    g1[5] = (int)ELEMS;                                        // tensor_dim0_stride lo32
    g1[6] = 0;
    g1[7] = 0;
    i32x4 z4 = {0, 0, 0, 0};
#if __clang_major__ >= 23
    i32x8 z8 = {0, 0, 0, 0, 0, 0, 0, 0};
    __builtin_amdgcn_tensor_load_to_lds(g0, g1, z4, z4, z8, 0);
#else
    __builtin_amdgcn_tensor_load_to_lds(g0, g1, z4, z4, 0);
#endif
    __builtin_amdgcn_s_wait_tensorcnt(0);
}
#endif

// ---------- 1) LayerNorm -> bf16 (one wave per 256-wide row) ----------
__global__ __launch_bounds__(256) void ln_to_bf16_kernel(
    const float* __restrict__ x, const float* __restrict__ g,
    const float* __restrict__ bb, unsigned short* __restrict__ out, int rows) {
    int wave = (blockIdx.x * blockDim.x + threadIdx.x) >> 5;
    int lane = threadIdx.x & 31;
    if (wave >= rows) return;
    const float4* xr = (const float4*)(x + (size_t)wave * 256);
    float4 a = xr[lane * 2 + 0];
    float4 b = xr[lane * 2 + 1];
    float xs[8] = {a.x, a.y, a.z, a.w, b.x, b.y, b.z, b.w};
    float s = 0.f;
    #pragma unroll
    for (int j = 0; j < 8; ++j) s += xs[j];
    float mu = waveSum(s) * (1.0f / 256.0f);
    float vs = 0.f;
    #pragma unroll
    for (int j = 0; j < 8; ++j) { float d = xs[j] - mu; vs += d * d; }
    float rstd = rsqrtf(waveSum(vs) * (1.0f / 256.0f) + 1e-5f);
    int c0 = lane * 8;
    unsigned int p[4];
    #pragma unroll
    for (int j = 0; j < 4; ++j) {
        float y0 = (xs[2 * j]     - mu) * rstd * g[c0 + 2 * j]     + bb[c0 + 2 * j];
        float y1 = (xs[2 * j + 1] - mu) * rstd * g[c0 + 2 * j + 1] + bb[c0 + 2 * j + 1];
        p[j] = pack2bf(y0, y1);
    }
    uint4 o; o.x = p[0]; o.y = p[1]; o.z = p[2]; o.w = p[3];
    ((uint4*)(out + (size_t)wave * 256))[lane] = o;
}

// ---------- 2) Bilinear grid sample -> bf16, layout (B,N,C,D) ----------
__global__ __launch_bounds__(256) void sample_kernel(
    const float* __restrict__ feat, const float* __restrict__ coords,
    unsigned short* __restrict__ outS) {
    constexpr int cC = 6, cN = 8192, cD = 256, cHf = 64, cWf = 64;
    int wave = (blockIdx.x * blockDim.x + threadIdx.x) >> 5;
    int lane = threadIdx.x & 31;
    int n  = wave % cN;
    int bc = wave / cN;            // bc = b*C + c
    int c  = bc % cC;
    int b  = bc / cC;
    float gx = coords[((size_t)bc * cN + n) * 2 + 0];
    float gy = coords[((size_t)bc * cN + n) * 2 + 1];
    float xf = (gx + 1.0f) * 0.5f * (cWf - 1);
    float yf = (gy + 1.0f) * 0.5f * (cHf - 1);
    float x0f = floorf(xf), y0f = floorf(yf);
    float wx1 = xf - x0f, wx0 = 1.f - wx1;
    float wy1 = yf - y0f, wy0 = 1.f - wy1;
    int x0 = (int)x0f, y0 = (int)y0f, x1 = x0 + 1, y1 = y0 + 1;
    auto inb = [](int yy, int xx) -> float {
        return (xx >= 0 && xx < cWf && yy >= 0 && yy < cHf) ? 1.f : 0.f;
    };
    float w00 = wy0 * wx0 * inb(y0, x0);
    float w01 = wy0 * wx1 * inb(y0, x1);
    float w10 = wy1 * wx0 * inb(y1, x0);
    float w11 = wy1 * wx1 * inb(y1, x1);
    auto clampc = [](int v, int hi) { return v < 0 ? 0 : (v > hi ? hi : v); };
    int xc0 = clampc(x0, cWf - 1), xc1 = clampc(x1, cWf - 1);
    int yc0 = clampc(y0, cHf - 1), yc1 = clampc(y1, cHf - 1);
    const float* base = feat + (size_t)bc * (cHf * cWf * cD);
    int e0 = lane * 8;
    float acc[8] = {};
    auto addc = [&](int yy, int xx, float w) {
        const float4* p = (const float4*)(base + ((size_t)yy * cWf + xx) * cD + e0);
        float4 u = p[0], v2 = p[1];
        acc[0] += w * u.x;  acc[1] += w * u.y;  acc[2] += w * u.z;  acc[3] += w * u.w;
        acc[4] += w * v2.x; acc[5] += w * v2.y; acc[6] += w * v2.z; acc[7] += w * v2.w;
    };
    addc(yc0, xc0, w00); addc(yc0, xc1, w01);
    addc(yc1, xc0, w10); addc(yc1, xc1, w11);
    uint4 o;
    o.x = pack2bf(acc[0], acc[1]); o.y = pack2bf(acc[2], acc[3]);
    o.z = pack2bf(acc[4], acc[5]); o.w = pack2bf(acc[6], acc[7]);
    ((uint4*)(outS + ((size_t)(b * cN + n) * cC + c) * cD))[lane] = o;
}

// ---------- 3) WMMA GEMM: out = A(bf16, M x 256) @ W(f32, 256 x Nc) + bias (+res) ----------
// Block = 128 thr = 4 waves. Block tile 32(M) x 64(N); wave nw owns N tile 16*nw and BOTH
// M strips (two accumulators sharing one B fragment -> 12 LDS dwords per WMMA, 2 chains).
// A tile staged by the Tensor Data Mover with DMA-side padding; W staged transposed f32->bf16.
template <int RESID, int OUTBF16>
__global__ __launch_bounds__(128) void gemm_bf16_wmma(
    const unsigned short* __restrict__ A, const float* __restrict__ Bw,
    const float* __restrict__ bias, const float* __restrict__ residual,
    void* __restrict__ out, int M, int Nc) {
    __shared__ uint4 ldsA4[32 * 33];   // 32 rows x 256 bf16, stride 132 dw
    __shared__ uint4 ldsB4[64 * 33];   // 64 n-rows x 256 bf16 transposed, stride 132 dw
    unsigned int* ldsA32 = (unsigned int*)ldsA4;
    unsigned int* ldsB32 = (unsigned int*)ldsB4;
    int t = threadIdx.x;
    int Mbase = blockIdx.x * 32;
    int Nbase = blockIdx.y * 64;
#if USE_TDM
    if (t < 32) {   // wave 0 issues one DMA for the whole 16 KiB A tile
        tdm_load_a_tile(A + (size_t)Mbase * 256, ldsA4);
    }
#else
    const uint4* Ag = (const uint4*)A;
    for (int i = t; i < 32 * 32; i += 128) {
        int row = i >> 5, c16 = i & 31;
        ldsA4[row * 33 + c16] = Ag[(size_t)(Mbase + row) * 32 + c16];
    }
#endif
    // stage W transposed as [n][k] bf16; t: n = t&63, k-range (t>>6)*128
    {
        int nloc = t & 63;
        int k0 = (t >> 6) * 128;
        const float* bp = Bw + (size_t)k0 * Nc + (Nbase + nloc);
        unsigned int* dst = ldsB32 + (size_t)nloc * 132 + (k0 >> 1);
        #pragma unroll 8
        for (int kk = 0; kk < 128; kk += 2) {
            float w0 = bp[(size_t)kk * Nc];
            float w1 = bp[(size_t)(kk + 1) * Nc];
            dst[kk >> 1] = pack2bf(w0, w1);
        }
    }
    __syncthreads();
    int nw = t >> 5, lane = t & 31;
    int l15 = lane & 15, hi = lane >> 4;
    const unsigned int* arow0 = ldsA32 + (size_t)l15 * 132;         // M strip 0
    const unsigned int* arow1 = ldsA32 + (size_t)(16 + l15) * 132;  // M strip 1
    const unsigned int* brow  = ldsB32 + (size_t)(16 * nw + l15) * 132;
    union FragBF { unsigned int u[8]; v16bf v; };
    FragBF a0, a1, bf;
    v8f acc0 = {}, acc1 = {};
    #pragma unroll
    for (int kb = 0; kb < 8; ++kb) {
        // A 16x32 frag: lane half hi -> K base +8; pairs {K..K+7} and {K+16..K+23}
        int ai = (kb * 32 + hi * 8) >> 1;
        #pragma unroll
        for (int j = 0; j < 4; ++j) { a0.u[j] = arow0[ai + j]; a0.u[4 + j] = arow0[ai + 8 + j]; }
        #pragma unroll
        for (int j = 0; j < 4; ++j) { a1.u[j] = arow1[ai + j]; a1.u[4 + j] = arow1[ai + 8 + j]; }
        // B 32x16 frag: lane half hi -> K base +8; pairs {K..K+7} and {K+16..K+23}
        int bi = (kb * 32 + hi * 8) >> 1;
        #pragma unroll
        for (int j = 0; j < 4; ++j) { bf.u[j] = brow[bi + j]; bf.u[4 + j] = brow[bi + 8 + j]; }
        acc0 = __builtin_amdgcn_wmma_f32_16x16x32_bf16(false, a0.v, false, bf.v,
                                                       (short)0, acc0, false, false);
        acc1 = __builtin_amdgcn_wmma_f32_16x16x32_bf16(false, a1.v, false, bf.v,
                                                       (short)0, acc1, false, false);
    }
    // C/D layout: lane n = l15, VGPR j -> M = j + 8*hi
    int gn = Nbase + 16 * nw + l15;
    float bv = bias[gn];
    #pragma unroll
    for (int j = 0; j < 8; ++j) {
        int gm0 = Mbase + hi * 8 + j;
        size_t o0 = (size_t)gm0 * Nc + gn;
        float v0 = acc0[j] + bv;
        if (RESID) v0 += residual[o0];
        if (OUTBF16) ((unsigned short*)out)[o0] = f2bf(v0);
        else         ((float*)out)[o0] = v0;
        int gm1 = Mbase + 16 + hi * 8 + j;
        size_t o1 = (size_t)gm1 * Nc + gn;
        float v1 = acc1[j] + bv;
        if (RESID) v1 += residual[o1];
        if (OUTBF16) ((unsigned short*)out)[o1] = f2bf(v1);
        else         ((float*)out)[o1] = v1;
    }
}

// ---------- 4) masked softmax attention over C=6, one wave per (b,n) ----------
__global__ __launch_bounds__(256) void attn_kernel(
    const float* __restrict__ q, const unsigned short* __restrict__ kv,
    const int* __restrict__ valid, unsigned short* __restrict__ outA) {
    constexpr int cB = 2, cC = 6, cN = 8192, cD = 256;
    int r = (blockIdx.x * blockDim.x + threadIdx.x) >> 5;   // r = b*N + n
    int lane = threadIdx.x & 31;
    if (r >= cB * cN) return;
    int b = r / cN, n = r % cN;
    int e0 = lane * 8;
    const float4* qp = (const float4*)(q + (size_t)r * cD + e0);
    float4 q0 = qp[0], q1 = qp[1];
    float qv[8] = {q0.x, q0.y, q0.z, q0.w, q1.x, q1.y, q1.z, q1.w};
    float lg[cC];
    #pragma unroll
    for (int c = 0; c < cC; ++c) {
        uint4 kk = ((const uint4*)(kv + ((size_t)r * cC + c) * 512 + e0))[0];
        unsigned int uu[4] = {kk.x, kk.y, kk.z, kk.w};
        float p = 0.f;
        #pragma unroll
        for (int j = 0; j < 4; ++j) {
            p += qv[2 * j]     * bf2f((unsigned short)(uu[j] & 0xffffu));
            p += qv[2 * j + 1] * bf2f((unsigned short)(uu[j] >> 16));
        }
        lg[c] = waveSum(p) * 0.0625f;   // / sqrt(256)
    }
    float vm[cC], mx = -1e30f;
    #pragma unroll
    for (int c = 0; c < cC; ++c) {
        int vl = valid[((size_t)(b * cC + c)) * cN + n];
        vm[c] = vl ? 1.f : 0.f;
        float cand = vl ? lg[c] : -1e30f;
        mx = cand > mx ? cand : mx;
    }
    float es[cC], ssum = 0.f;
    #pragma unroll
    for (int c = 0; c < cC; ++c) {
        es[c] = vm[c] != 0.f ? expf(lg[c] - mx) : 0.f;
        ssum += es[c];
    }
    float inv = 1.0f / (ssum == 0.f ? 1.f : ssum);
    float acc[8] = {};
    #pragma unroll
    for (int c = 0; c < cC; ++c) {
        float a = es[c] * inv;
        uint4 vv = ((const uint4*)(kv + ((size_t)r * cC + c) * 512 + 256 + e0))[0];
        unsigned int uu[4] = {vv.x, vv.y, vv.z, vv.w};
        #pragma unroll
        for (int j = 0; j < 4; ++j) {
            acc[2 * j]     += a * bf2f((unsigned short)(uu[j] & 0xffffu));
            acc[2 * j + 1] += a * bf2f((unsigned short)(uu[j] >> 16));
        }
    }
    uint4 o;
    o.x = pack2bf(acc[0], acc[1]); o.y = pack2bf(acc[2], acc[3]);
    o.z = pack2bf(acc[4], acc[5]); o.w = pack2bf(acc[6], acc[7]);
    ((uint4*)(outA + (size_t)r * cD))[lane] = o;
}

extern "C" void kernel_launch(void* const* d_in, const int* in_sizes, int n_in,
                              void* d_out, int out_size, void* d_ws, size_t ws_size,
                              hipStream_t stream) {
    const float* queries        = (const float*)d_in[0];
    const float* image_features = (const float*)d_in[1];
    const float* pixel_coords   = (const float*)d_in[2];
    const int*   valid_mask     = (const int*)d_in[3];
    const float* q_w   = (const float*)d_in[4];
    const float* q_b   = (const float*)d_in[5];
    const float* kv_w  = (const float*)d_in[6];
    const float* kv_b  = (const float*)d_in[7];
    const float* out_w = (const float*)d_in[8];
    const float* out_b = (const float*)d_in[9];
    const float* ln_g  = (const float*)d_in[10];
    const float* ln_b  = (const float*)d_in[11];

    constexpr size_t BN  = 2 * 8192;    // 16384 query rows
    constexpr size_t BNC = BN * 6;      // 98304 sampled rows
    char* ws = (char*)d_ws;
    size_t off = 0;
    unsigned short* qn    = (unsigned short*)(ws + off); off += BN  * 256 * 2; //  8 MiB
    unsigned short* samp  = (unsigned short*)(ws + off); off += BNC * 256 * 2; // 48 MiB
    unsigned short* kvb   = (unsigned short*)(ws + off); off += BNC * 512 * 2; // 96 MiB
    float*          qf    = (float*)(ws + off);          off += BN  * 256 * 4; // 16 MiB
    unsigned short* attnb = (unsigned short*)(ws + off); off += BN  * 256 * 2; //  8 MiB
    (void)ws_size; (void)in_sizes; (void)n_in; (void)out_size;

    // 1) layernorm -> qn (bf16)
    ln_to_bf16_kernel<<<BN / 8, 256, 0, stream>>>(queries, ln_g, ln_b, qn, (int)BN);
    // 2) bilinear sample -> samp (bf16, (B,N,C,D))
    sample_kernel<<<BNC / 8, 256, 0, stream>>>(image_features, pixel_coords, samp);
    // 3) q = qn @ q_w + q_b (fp32 out)
    gemm_bf16_wmma<0, 0><<<dim3(BN / 32, 4), 128, 0, stream>>>(qn, q_w, q_b, nullptr, qf,
                                                               (int)BN, 256);
    // 4) kv = samp @ kv_w + kv_b (bf16 out)
    gemm_bf16_wmma<0, 1><<<dim3(BNC / 32, 8), 128, 0, stream>>>(samp, kv_w, kv_b, nullptr, kvb,
                                                                (int)BNC, 512);
    // 5) masked softmax attention -> attnb (bf16)
    attn_kernel<<<BN / 8, 256, 0, stream>>>(qf, kvb, valid_mask, attnb);
    // 6) out = residual + attnb @ out_w + out_b (fp32 -> d_out)
    gemm_bf16_wmma<1, 0><<<dim3(BN / 32, 4), 128, 0, stream>>>(attnb, out_w, out_b, queries,
                                                               (float*)d_out, (int)BN, 256);
}
